// GCNDecoder_38689065402409
// MI455X (gfx1250) — matrix operand order repaired and checked by reading, not compiled, not run on hardware
//
#include <hip/hip_runtime.h>

// ---------------------------------------------------------------------------
// GCN decoder for MI455X (gfx1250, wave32, WMMA).
// s = A_norm^T x   (bf16 WMMA, K=2048, adj normalized+transposed on the fly)
// then fused: h2 = s@WcT + cb ; h3 = h2@WmT + mb ; x = relu(LN(h3));
// last layer additionally fuses mu = x@linT + lin_b -> d_out (f32).
// Pure global->LDS staging copies use GLOBAL_LOAD_ASYNC_TO_LDS_B128 when the
// toolchain exposes the gfx1250 async builtins (falls back to b128 copies).
// ---------------------------------------------------------------------------

typedef __attribute__((ext_vector_type(16))) __bf16 v16bf;
typedef __attribute__((ext_vector_type(8)))  __bf16 v8bf;
typedef __attribute__((ext_vector_type(8)))  float  v8f;

#define G_   16
#define N_   2048
#define H_   128
#define OUT_ 64

#if defined(__HIP_DEVICE_COMPILE__) && defined(__has_builtin)
#if __has_builtin(__builtin_amdgcn_global_load_async_to_lds_b128)
#define HAVE_ASYNC_LDS 1
#endif
#endif
#ifndef HAVE_ASYNC_LDS
#define HAVE_ASYNC_LDS 0
#endif

// 16-byte global -> LDS copy; async DMA (ASYNCcnt) on CDNA5 toolchains.
// ROCm 7.2 signature (leaked by diagnostics): (AS1 v4i*, AS3 v4i*, Ii, Ii).
__device__ __forceinline__ void cp16_g2l(const void* g, void* l) {
#if HAVE_ASYNC_LDS
  typedef int v4i_ __attribute__((vector_size(16)));
  __builtin_amdgcn_global_load_async_to_lds_b128(
      (__attribute__((address_space(1))) v4i_*)(unsigned long long)(size_t)g,
      (__attribute__((address_space(3))) v4i_*)(unsigned)(size_t)l,
      0, 0);
#else
  *(uint4*)l = *(const uint4*)g;
#endif
}

__device__ __forceinline__ void cp_g2l_wait() {
#if HAVE_ASYNC_LDS
#if __has_builtin(__builtin_amdgcn_s_wait_asynccnt)
  __builtin_amdgcn_s_wait_asynccnt(0);
#else
  asm volatile("s_wait_asynccnt 0" ::: "memory");
#endif
#endif
}

// Build a v16bf WMMA operand from two 16-byte LDS/global chunks.
__device__ __forceinline__ v16bf ldfrag(const __bf16* p0, const __bf16* p1) {
  v8bf lo = *(const v8bf*)p0;
  v8bf hi = *(const v8bf*)p1;
  return __builtin_shufflevector(lo, hi, 0,1,2,3,4,5,6,7,8,9,10,11,12,13,14,15);
}

__device__ __forceinline__ v8f bmma(v16bf a, v16bf b, v8f c) {
  return __builtin_amdgcn_wmma_f32_16x16x32_bf16(false, a, false, b,
                                                 (short)0, c, false, false);
}

__device__ __forceinline__ unsigned pk2(const __bf16* p0, const __bf16* p1) {
  unsigned a = (unsigned)__builtin_bit_cast(unsigned short, *p0);
  unsigned b = (unsigned)__builtin_bit_cast(unsigned short, *p1);
  return a | (b << 16);
}

// --------------------------- degree / rsqrt --------------------------------
// d_is[g,j] = rsqrt(1 + sum_i adj[g,i,j])   (self loop makes deg >= 1)
__global__ __launch_bounds__(256) void kdeg(const float* __restrict__ adj,
                                            float* __restrict__ d_is) {
  int id = blockIdx.x * 256 + threadIdx.x;      // g*N + j
  int g = id >> 11, j = id & (N_ - 1);
  const float* c = adj + (size_t)g * N_ * N_ + j;
  float s = 1.0f;
  for (int i = 0; i < N_; i += 4) {
    s += c[(size_t)i * N_] + c[(size_t)(i + 1) * N_] +
         c[(size_t)(i + 2) * N_] + c[(size_t)(i + 3) * N_];
  }
  d_is[id] = rsqrtf(s);
}

// ------------------------- weight transposes -------------------------------
// WcT/WmT: [l][k][h] = w[l][h][k] (bf16);  linT: [o][h] = lin_w[h][o] (bf16)
__global__ __launch_bounds__(256) void kweights(const float* __restrict__ cw,
                                                const float* __restrict__ mw,
                                                const float* __restrict__ lw,
                                                __bf16* __restrict__ wcT,
                                                __bf16* __restrict__ wmT,
                                                __bf16* __restrict__ linT) {
  int idx = blockIdx.x * 256 + threadIdx.x;
  const int nw = 3 * H_ * H_;
  if (idx < nw) {
    int l = idx / (H_ * H_), r = idx % (H_ * H_), k = r >> 7, h = r & 127;
    wcT[idx] = (__bf16)cw[l * H_ * H_ + h * H_ + k];
  } else if (idx < 2 * nw) {
    int j = idx - nw;
    int l = j / (H_ * H_), r = j % (H_ * H_), k = r >> 7, h = r & 127;
    wmT[j] = (__bf16)mw[l * H_ * H_ + h * H_ + k];
  } else if (idx < 2 * nw + OUT_ * H_) {
    int j = idx - 2 * nw, o = j >> 7, h = j & 127;
    linT[j] = (__bf16)lw[h * OUT_ + o];
  }
}

// ------------------- initial feature transpose to bf16 ---------------------
// xT[g][h][i] = bf16(node_feat[g][i][h]);  32x32 LDS tile transpose
__global__ __launch_bounds__(256) void kx0(const float* __restrict__ nf,
                                           __bf16* __restrict__ xT) {
  __shared__ __bf16 tile[32][33];
  int g = blockIdx.z, ib = blockIdx.x * 32, hb = blockIdx.y * 32;
  int tx = threadIdx.x, ty = threadIdx.y;     // 32 x 8
#pragma unroll
  for (int k = 0; k < 32; k += 8)
    tile[ty + k][tx] = (__bf16)nf[((size_t)g * N_ + ib + ty + k) * H_ + hb + tx];
  __syncthreads();
#pragma unroll
  for (int k = 0; k < 32; k += 8)
    xT[((size_t)g * H_ + hb + ty + k) * N_ + ib + tx] = tile[tx][ty + k];
}

// ----------------------------- aggregation ---------------------------------
// s[g][j][h] = sum_i d_i d_j (adj[g][i][j] + [i==j]) * x[g][i][h]
// Block: 256 thr (8 waves, 4m x 2n), tile 128(j) x 128(h), K-step 64 over i.
// A staged transposed+normalized bf16 in LDS; B async-DMA'd from xT.
#define LDA 72   // 64 + 8 halves pad -> conflict-free b128 frag loads
__global__ __launch_bounds__(256) void kagg(const float* __restrict__ adj,
                                            const float* __restrict__ dis,
                                            const __bf16* __restrict__ xT,
                                            __bf16* __restrict__ s) {
  __shared__ __bf16 Al[128 * LDA];
  __shared__ __bf16 Bl[128 * LDA];
  __shared__ float dall[N_];

  int g = blockIdx.y;
  int j0 = blockIdx.x * 128;
  int t = threadIdx.x;
  int lane = t & 31, wid = t >> 5;
  int wm = wid & 3, wn = wid >> 2;      // wave tile: 32(j) x 64(h)
  int half = lane >> 4, l16 = lane & 15;

  const float* adjg = adj + (size_t)g * N_ * N_;
  const __bf16* xTg = xT + (size_t)g * H_ * N_;

#pragma unroll
  for (int q = 0; q < 8; ++q) dall[t + q * 256] = dis[g * N_ + t + q * 256];

  v8f acc[2][4] = {};
  __syncthreads();

  for (int i0 = 0; i0 < N_; i0 += 64) {
    // stage B first: async DMA xT[h][i0..+63] straight into LDS (ASYNCcnt)
#pragma unroll
    for (int q = 0; q < 4; ++q) {
      int idx = t + q * 256;                 // 1024 uint4 chunks
      int h = idx >> 3, iseg = idx & 7;
      cp16_g2l(xTg + (size_t)h * N_ + i0 + iseg * 8, &Bl[h * LDA + iseg * 8]);
    }
    // stage A: adj[i0..+63][j0..+127] -> normalized bf16, transposed [j][i]
#pragma unroll
    for (int q = 0; q < 8; ++q) {
      int idx = t + q * 256;                 // 2048 float4 chunks
      int ii = idx >> 5, jseg = idx & 31;
      int i = i0 + ii, jb = j0 + jseg * 4;
      float4 v = *(const float4*)(adjg + (size_t)i * N_ + jb);
      float di = dall[i];
      float f0 = (v.x + ((i == jb + 0) ? 1.0f : 0.0f)) * di * dall[jb + 0];
      float f1 = (v.y + ((i == jb + 1) ? 1.0f : 0.0f)) * di * dall[jb + 1];
      float f2 = (v.z + ((i == jb + 2) ? 1.0f : 0.0f)) * di * dall[jb + 2];
      float f3 = (v.w + ((i == jb + 3) ? 1.0f : 0.0f)) * di * dall[jb + 3];
      Al[(jseg * 4 + 0) * LDA + ii] = (__bf16)f0;
      Al[(jseg * 4 + 1) * LDA + ii] = (__bf16)f1;
      Al[(jseg * 4 + 2) * LDA + ii] = (__bf16)f2;
      Al[(jseg * 4 + 3) * LDA + ii] = (__bf16)f3;
    }
    // prefetch next adj tile toward GL2 while this tile computes
    if (i0 + 64 < N_) {
      int idx = t, ii = idx >> 5, jseg = idx & 31;
      __builtin_prefetch(adjg + (size_t)(i0 + 64 + ii) * N_ + j0 + jseg * 4, 0, 0);
      __builtin_prefetch(adjg + (size_t)(i0 + 96 + ii) * N_ + j0 + jseg * 4, 0, 0);
    }
    cp_g2l_wait();
    __syncthreads();

#pragma unroll
    for (int kc = 0; kc < 2; ++kc) {         // two K=32 chunks per tile
      v16bf a[2], b[4];
#pragma unroll
      for (int m = 0; m < 2; ++m) {
        int row = wm * 32 + m * 16 + l16;
        int kb = half * 16 + kc * 32;
        a[m] = ldfrag(&Al[row * LDA + kb], &Al[row * LDA + kb + 8]);
      }
#pragma unroll
      for (int n = 0; n < 4; ++n) {
        int row = wn * 64 + n * 16 + l16;
        int kk = half * 16 + kc * 32;
        b[n] = ldfrag(&Bl[row * LDA + kk], &Bl[row * LDA + kk + 8]);
      }
#pragma unroll
      for (int m = 0; m < 2; ++m)
#pragma unroll
        for (int n = 0; n < 4; ++n)
          acc[m][n] = bmma(a[m], b[n], acc[m][n]);
    }
    __syncthreads();
  }

  __bf16* sg = s + ((size_t)g * N_ + j0) * H_;
#pragma unroll
  for (int m = 0; m < 2; ++m)
#pragma unroll
    for (int n = 0; n < 4; ++n)
#pragma unroll
      for (int r = 0; r < 8; ++r) {
        int j = wm * 32 + m * 16 + r + 8 * half;
        int h = wn * 64 + n * 16 + l16;
        sg[(size_t)j * H_ + h] = (__bf16)acc[m][n][r];
      }
}

// ------------------- fused MLP + LayerNorm + ReLU (+ head) -----------------
// Block: 256 thr, 128 rows; each wave owns 16 rows (all 128 cols).
#define LDS 136  // 128 + 8 halves pad
__global__ __launch_bounds__(256) void kmlp(
    const __bf16* __restrict__ s, const __bf16* __restrict__ wcT,
    const float* __restrict__ cb, const __bf16* __restrict__ wmT,
    const float* __restrict__ mb, const float* __restrict__ lng,
    const float* __restrict__ lnb, const __bf16* __restrict__ linT,
    const float* __restrict__ lb, __bf16* __restrict__ xT,
    float* __restrict__ out, int last) {
  __shared__ __bf16 sx[128 * LDS];
  int g = blockIdx.y, i0 = blockIdx.x * 128;
  int t = threadIdx.x, lane = t & 31, wid = t >> 5;
  int half = lane >> 4, l16 = lane & 15;
  int rbase = wid * 16;

  const __bf16* sgp = s + ((size_t)g * N_ + i0) * H_;
#pragma unroll
  for (int q = 0; q < 8; ++q) {
    int idx = t + q * 256;                   // 2048 uint4 chunks
    int r = idx >> 4, cseg = idx & 15;
    cp16_g2l(sgp + (size_t)r * H_ + cseg * 8, &sx[r * LDS + cseg * 8]);
  }
  cp_g2l_wait();
  __syncthreads();

  // GEMM1: h2 = s @ WcT  (+cb), result back into own LDS rows as bf16
  v8f acc1[8] = {};
#pragma unroll
  for (int ks = 0; ks < 4; ++ks) {
    int kb = half * 16 + ks * 32;
    int arow = rbase + l16;
    v16bf a = ldfrag(&sx[arow * LDS + kb], &sx[arow * LDS + kb + 8]);
#pragma unroll
    for (int n = 0; n < 8; ++n) {
      const __bf16* wr = wcT + (n * 16 + l16) * H_ + half * 16 + ks * 32;
      acc1[n] = bmma(a, ldfrag(wr, wr + 8), acc1[n]);
    }
  }
#pragma unroll
  for (int n = 0; n < 8; ++n) {
    float bv = cb[n * 16 + l16];
#pragma unroll
    for (int r = 0; r < 8; ++r)
      sx[(rbase + r + 8 * half) * LDS + n * 16 + l16] = (__bf16)(acc1[n][r] + bv);
  }
  asm volatile("s_wait_dscnt 0" ::: "memory");

  // GEMM2: h3 = h2 @ WmT (+mb)
  v8f acc2[8] = {};
#pragma unroll
  for (int ks = 0; ks < 4; ++ks) {
    int kb = half * 16 + ks * 32;
    int arow = rbase + l16;
    v16bf a = ldfrag(&sx[arow * LDS + kb], &sx[arow * LDS + kb + 8]);
#pragma unroll
    for (int n = 0; n < 8; ++n) {
      const __bf16* wr = wmT + (n * 16 + l16) * H_ + half * 16 + ks * 32;
      acc2[n] = bmma(a, ldfrag(wr, wr + 8), acc2[n]);
    }
  }
#pragma unroll
  for (int n = 0; n < 8; ++n) {
    float bv = mb[n * 16 + l16];
#pragma unroll
    for (int r = 0; r < 8; ++r) acc2[n][r] += bv;
  }

  // LayerNorm over H=128: row stats via 16-lane shuffle reduce (wave32 halves)
  float mu[8], rs[8];
#pragma unroll
  for (int r = 0; r < 8; ++r) {
    float s1 = 0.f, s2 = 0.f;
#pragma unroll
    for (int n = 0; n < 8; ++n) { float v = acc2[n][r]; s1 += v; s2 += v * v; }
#pragma unroll
    for (int m = 1; m < 16; m <<= 1) {
      s1 += __shfl_xor(s1, m, 32);
      s2 += __shfl_xor(s2, m, 32);
    }
    float me = s1 * (1.0f / H_);
    mu[r] = me;
    rs[r] = rsqrtf(fmaxf(s2 * (1.0f / H_) - me * me, 0.f) + 1e-5f);
  }
#pragma unroll
  for (int n = 0; n < 8; ++n) {
    int col = n * 16 + l16;
    float gv = lng[col], bv = lnb[col];
#pragma unroll
    for (int r = 0; r < 8; ++r) {
      float v = (acc2[n][r] - mu[r]) * rs[r] * gv + bv;
      sx[(rbase + r + 8 * half) * LDS + col] = (__bf16)fmaxf(v, 0.f);
    }
  }
  asm volatile("s_wait_dscnt 0" ::: "memory");

  if (last) {
    // head: mu = x @ linT + lin_b -> d_out (f32)
    v8f acc3[4] = {};
#pragma unroll
    for (int ks = 0; ks < 4; ++ks) {
      int kb = half * 16 + ks * 32;
      int arow = rbase + l16;
      v16bf a = ldfrag(&sx[arow * LDS + kb], &sx[arow * LDS + kb + 8]);
#pragma unroll
      for (int n = 0; n < 4; ++n) {
        const __bf16* wr = linT + (n * 16 + l16) * H_ + half * 16 + ks * 32;
        acc3[n] = bmma(a, ldfrag(wr, wr + 8), acc3[n]);
      }
    }
#pragma unroll
    for (int n = 0; n < 4; ++n) {
      float b = lb[n * 16 + l16];
#pragma unroll
      for (int r = 0; r < 8; ++r) {
        int i = i0 + rbase + r + 8 * half;
        out[((size_t)g * N_ + i) * OUT_ + n * 16 + l16] = acc3[n][r] + b;
      }
    }
  } else {
    // transposed bf16 write-back: xT[g][h][i0..+127]
    __syncthreads();
    int h = t >> 1, seg = t & 1;
    __bf16* dst = xT + ((size_t)g * H_ + h) * N_ + i0 + seg * 64;
#pragma unroll
    for (int q = 0; q < 8; ++q) {
      int ib = seg * 64 + q * 8;
      uint4 w;
      w.x = pk2(&sx[(ib + 0) * LDS + h], &sx[(ib + 1) * LDS + h]);
      w.y = pk2(&sx[(ib + 2) * LDS + h], &sx[(ib + 3) * LDS + h]);
      w.z = pk2(&sx[(ib + 4) * LDS + h], &sx[(ib + 5) * LDS + h]);
      w.w = pk2(&sx[(ib + 6) * LDS + h], &sx[(ib + 7) * LDS + h]);
      *(uint4*)(dst + q * 8) = w;
    }
  }
}

// ---------------------------------------------------------------------------
extern "C" void kernel_launch(void* const* d_in, const int* in_sizes, int n_in,
                              void* d_out, int out_size, void* d_ws,
                              size_t ws_size, hipStream_t stream) {
  (void)in_sizes; (void)n_in; (void)out_size; (void)ws_size;
  const float* node_feat = (const float*)d_in[0];
  const float* adj       = (const float*)d_in[1];
  const float* conv_w    = (const float*)d_in[2];
  const float* conv_b    = (const float*)d_in[3];
  const float* mlp_w     = (const float*)d_in[4];
  const float* mlp_b     = (const float*)d_in[5];
  const float* ln_g      = (const float*)d_in[6];
  const float* ln_b      = (const float*)d_in[7];
  const float* lin_w     = (const float*)d_in[8];
  const float* lin_b     = (const float*)d_in[9];
  float* out = (float*)d_out;

  char* ws = (char*)d_ws;
  size_t off = 0;
  float* d_is = (float*)(ws + off);           off += (size_t)G_ * N_ * 4;
  __bf16* xT  = (__bf16*)(ws + off);          off += (size_t)G_ * H_ * N_ * 2;
  __bf16* sbf = (__bf16*)(ws + off);          off += (size_t)G_ * N_ * H_ * 2;
  __bf16* wcT = (__bf16*)(ws + off);          off += (size_t)3 * H_ * H_ * 2;
  __bf16* wmT = (__bf16*)(ws + off);          off += (size_t)3 * H_ * H_ * 2;
  __bf16* linT = (__bf16*)(ws + off);         off += (size_t)OUT_ * H_ * 2;

  int nwt = 2 * 3 * H_ * H_ + OUT_ * H_;
  kweights<<<(nwt + 255) / 256, 256, 0, stream>>>(conv_w, mlp_w, lin_w,
                                                  wcT, wmT, linT);
  kdeg<<<(G_ * N_) / 256, 256, 0, stream>>>(adj, d_is);
  kx0<<<dim3(N_ / 32, H_ / 32, G_), dim3(32, 8), 0, stream>>>(node_feat, xT);

  for (int l = 0; l < 3; ++l) {
    kagg<<<dim3(N_ / 128, G_), 256, 0, stream>>>(adj, d_is, xT, sbf);
    kmlp<<<dim3(N_ / 128, G_), 256, 0, stream>>>(
        sbf, wcT + (size_t)l * H_ * H_, conv_b + l * H_,
        wmT + (size_t)l * H_ * H_, mlp_b + l * H_, ln_g + l * H_,
        ln_b + l * H_, linT, lin_b, xT, out, (l == 2) ? 1 : 0);
  }
}
